// Decoder_86732569575653
// MI455X (gfx1250) — matrix operand order repaired and checked by reading, not compiled, not run on hardware
//
#include <hip/hip_runtime.h>
#include <hip/hip_bf16.h>
#include <math.h>

// Problem dims (fixed by reference)
constexpr int V = 64, E = 256, L = 256, C = 64, H = 1024, B = 256, T = 256;
constexpr int G3H = 3 * H;          // 3072
constexpr int LC = L + C;           // 320
constexpr int ELC = E + L + C;      // 576

typedef __bf16 bf16;
typedef __attribute__((ext_vector_type(16))) __bf16 v16bf;
typedef __attribute__((ext_vector_type(8)))  __bf16 v8bf;
typedef __attribute__((ext_vector_type(8)))  float  v8f;

// ---------------------------------------------------------------------------
// Workspace layout (bytes, 256-aligned)
// ---------------------------------------------------------------------------
constexpr size_t OFF_H0    = 0;                         // [B,H] f32       1,048,576
constexpr size_t OFF_GTAB  = OFF_H0   + (size_t)B*H*4;  // [V,3H] f32        786,432
constexpr size_t OFF_GCTX  = OFF_GTAB + (size_t)V*G3H*4;// [B,3H] f32      3,145,728
constexpr size_t OFF_WHH   = OFF_GCTX + (size_t)B*G3H*4;// packed bf16     6,291,456
constexpr size_t OFF_WOUT  = OFF_WHH  + (size_t)(G3H/16)*(H/32)*512*2; //    131,072

// ---------------------------------------------------------------------------
// Setup kernels (tiny vs. the scan; plain VALU)
// ---------------------------------------------------------------------------

// h0[b,j] = context[b] . W_fch[j] + b_fch[j]
__global__ void k_h0(const float* __restrict__ z, const float* __restrict__ c,
                     const float* __restrict__ Wf, const float* __restrict__ bf,
                     float* __restrict__ h0f) {
    int idx = blockIdx.x * blockDim.x + threadIdx.x;   // b*H + j
    int b = idx >> 10, j = idx & (H - 1);
    const float* wr = Wf + (size_t)j * LC;
    float acc = bf[j];
    for (int i = 0; i < L; ++i) acc += z[b * L + i] * wr[i];
    for (int i = 0; i < C; ++i) acc += c[b * C + i] * wr[L + i];
    h0f[idx] = acc;
}

// gtab[v,g] = emb[v] . W_ih[g, 0:E]
__global__ void k_gtab(const float* __restrict__ emb, const float* __restrict__ Wih,
                       float* __restrict__ gtab) {
    int idx = blockIdx.x * blockDim.x + threadIdx.x;   // v*3H + g
    int v = idx / G3H, g = idx % G3H;
    const float* wr = Wih + (size_t)g * ELC;
    float acc = 0.f;
    for (int e = 0; e < E; ++e) acc += emb[v * E + e] * wr[e];
    gtab[idx] = acc;
}

// gctx[b,g] = context[b] . W_ih[g, E:] + b_ih[g]
__global__ void k_gctx(const float* __restrict__ z, const float* __restrict__ c,
                       const float* __restrict__ Wih, const float* __restrict__ bih,
                       float* __restrict__ gctx) {
    int idx = blockIdx.x * blockDim.x + threadIdx.x;   // b*3H + g
    int b = idx / G3H, g = idx % G3H;
    const float* wr = Wih + (size_t)g * ELC + E;
    float acc = bih[g];
    for (int i = 0; i < L; ++i) acc += z[b * L + i] * wr[i];
    for (int i = 0; i < C; ++i) acc += c[b * C + i] * wr[L + i];
    gctx[idx] = acc;
}

// Pack a row-major f32 weight [Nrows x H] into bf16 WMMA B fragments.
// Fragment layout per (nb, kb) tile: 32 lanes x 16 bf16 contiguous.
// Lane l, slot s: hi=l>>4; kk = (s<8) ? 8*hi+s : 16+8*hi+(s-8); nn = l&15.
__global__ void k_pack(const float* __restrict__ Wsrc, bf16* __restrict__ dst,
                       int kblocks /* H/32 */) {
    int idx = blockIdx.x * blockDim.x + threadIdx.x;
    int s    = idx & 15;
    int lane = (idx >> 4) & 31;
    int kb   = (idx >> 9) % kblocks;
    int nb   = (idx >> 9) / kblocks;
    int vv = s >> 1, p = s & 1, hi = lane >> 4;
    int kk = (s < 8) ? (8 * hi + s) : (16 + 8 * hi + (s - 8));
    int nn = lane & 15;
    int g = nb * 16 + nn;
    int k = kb * 32 + kk;
    dst[idx] = (bf16)Wsrc[(size_t)g * H + k];
}

// ---------------------------------------------------------------------------
// Persistent GRU scan + fused logits
// ---------------------------------------------------------------------------
#define NWAVES 32
#define TPB (NWAVES * 32)

// A-fragment (16x32 bf16, 16-bit A layout): lane holds row m=lane&15,
// two contiguous 8-element chunks: k = kb*32 + 8*hi + [0..7] and +16.
__device__ __forceinline__ v16bf load_a_frag(const bf16* hB, int kb, int lane) {
    int m = lane & 15, hi = lane >> 4;
    const bf16* base = hB + m * H + kb * 32 + 8 * hi;
    v8bf lo = *(const v8bf*)(base);
    v8bf up = *(const v8bf*)(base + 16);
    v16bf a;
#pragma unroll
    for (int i = 0; i < 8; ++i) { a[i] = lo[i]; a[i + 8] = up[i]; }
    return a;
}

__device__ __forceinline__ float sigmf(float x) { return 1.f / (1.f + __expf(-x)); }

__launch_bounds__(TPB, 1)
__global__ void gru_scan(const float* __restrict__ h0f,
                         const float* __restrict__ gtab,
                         const float* __restrict__ gctx,
                         const bf16*  __restrict__ whh_pk,
                         const bf16*  __restrict__ wout_pk,
                         const int*   __restrict__ seq,
                         const float* __restrict__ b_hh,
                         const float* __restrict__ b_out,
                         float*       __restrict__ out /* [B,T,V] */) {
    __shared__ bf16 hB[16 * H];     // 32 KB: this tile's hidden state, bf16
    __shared__ int  tokS[16];

    const int wg   = blockIdx.x;          // batch tile: rows wg*16 .. wg*16+15
    const int tid  = threadIdx.x;
    const int w    = tid >> 5;            // wave id 0..31
    const int lane = tid & 31;
    const int hi   = lane >> 4;
    const int nn   = lane & 15;

    // Load h0 tile into LDS (bf16)
    for (int i = tid; i < 16 * H; i += TPB)
        hB[i] = (bf16)h0f[(size_t)(wg * 16 + (i >> 10)) * H + (i & (H - 1))];
    __syncthreads();

    for (int t = 0; t < T; ++t) {
        if (tid < 16) tokS[tid] = seq[(wg * 16 + tid) * T + t];
        __syncthreads();

        // ---- gh = h @ W_hh^T : wave w owns columns [w*32, w*32+32).
        // Two k-passes of 3 output tiles (one per gate third) keep VGPR
        // pressure low enough for spill-free software pipelining at
        // 8 waves/SIMD. Pass s==0 on waves 0..3 also carries the logits
        // GEMM for step t-1 (same A fragments: logits[t-1] uses S_t);
        // the condition is hoisted out of the k-loop so each variant
        // pipelines with unconditional load->wmma chains.
        float hnew[2][8];
#pragma unroll
        for (int s = 0; s < 2; ++s) {
            const int nb0 = w * 2 + s;            // column tile within a gate third
            v8f acc[3];
#pragma unroll
            for (int i = 0; i < 3; ++i) acc[i] = (v8f){0,0,0,0,0,0,0,0};
            v8f lacc = (v8f){0,0,0,0,0,0,0,0};
            const bool do_log = (s == 0) && (w < 4);

            if (do_log) {
                // 4-tile body: 3 gate tiles + 1 logits tile share the A fragment
                for (int kb = 0; kb < H / 32; ++kb) {
                    v16bf bfr[3];
#pragma unroll
                    for (int cg = 0; cg < 3; ++cg) {
                        size_t nb = (size_t)(cg * (H / 16) + nb0);
                        bfr[cg] = *(const v16bf*)(whh_pk + (nb * (H / 32) + kb) * 512 + lane * 16);
                    }
                    const v16bf bfv = *(const v16bf*)(wout_pk + ((size_t)w * (H / 32) + kb) * 512 + lane * 16);
                    if (kb + 1 < H / 32)
                        __builtin_prefetch(whh_pk + ((size_t)nb0 * (H / 32) + kb + 1) * 512, 0, 3);
                    v16bf a = load_a_frag(hB, kb, lane);
#pragma unroll
                    for (int cg = 0; cg < 3; ++cg)
                        acc[cg] = __builtin_amdgcn_wmma_f32_16x16x32_bf16(
                            false, a, false, bfr[cg], (short)0, acc[cg], false, false);
                    lacc = __builtin_amdgcn_wmma_f32_16x16x32_bf16(
                        false, a, false, bfv, (short)0, lacc, false, false);
                }
            } else {
                // 3-tile body
                for (int kb = 0; kb < H / 32; ++kb) {
                    v16bf bfr[3];
#pragma unroll
                    for (int cg = 0; cg < 3; ++cg) {
                        size_t nb = (size_t)(cg * (H / 16) + nb0);
                        bfr[cg] = *(const v16bf*)(whh_pk + (nb * (H / 32) + kb) * 512 + lane * 16);
                    }
                    if (kb + 1 < H / 32)
                        __builtin_prefetch(whh_pk + ((size_t)nb0 * (H / 32) + kb + 1) * 512, 0, 3);
                    v16bf a = load_a_frag(hB, kb, lane);
#pragma unroll
                    for (int cg = 0; cg < 3; ++cg)
                        acc[cg] = __builtin_amdgcn_wmma_f32_16x16x32_bf16(
                            false, a, false, bfr[cg], (short)0, acc[cg], false, false);
                }
            }

            // deferred logits store for step t-1 (uses S_t = current hB)
            if (do_log && t > 0) {
                int vcol = w * 16 + nn;
                float bo = b_out[vcol];
#pragma unroll
                for (int r = 0; r < 8; ++r) {
                    int m  = r + 8 * hi;
                    int bg = wg * 16 + m;
                    out[((size_t)bg * T + (t - 1)) * V + vcol] = lacc[r] + bo;
                }
            }

            // gates for this pass's 16 hidden columns.
            // D layout: vgpr r, lane -> (m = r + 8*hi, n = lane&15)
            int j = w * 32 + s * 16 + nn;
            float bhR = b_hh[j], bhZ = b_hh[H + j], bhN = b_hh[2 * H + j];
#pragma unroll
            for (int r = 0; r < 8; ++r) {
                int m  = r + 8 * hi;
                int tk = tokS[m];
                int bg = wg * 16 + m;
                float giR = gtab[(size_t)tk * G3H + j]          + gctx[(size_t)bg * G3H + j];
                float giZ = gtab[(size_t)tk * G3H + H + j]      + gctx[(size_t)bg * G3H + H + j];
                float giN = gtab[(size_t)tk * G3H + 2 * H + j]  + gctx[(size_t)bg * G3H + 2 * H + j];
                float ghR = acc[0][r] + bhR;
                float ghZ = acc[1][r] + bhZ;
                float ghN = acc[2][r] + bhN;
                float rg = sigmf(giR + ghR);
                float u  = sigmf(giZ + ghZ);
                float nv = tanhf(giN + rg * ghN);
                float ho = (float)hB[m * H + j];
                hnew[s][r] = (1.f - u) * nv + u * ho;
            }
        }
        __syncthreads();   // all waves done reading old h
#pragma unroll
        for (int s = 0; s < 2; ++s) {
            int j = w * 32 + s * 16 + nn;
#pragma unroll
            for (int r = 0; r < 8; ++r) {
                int m = r + 8 * hi;
                hB[m * H + j] = (bf16)hnew[s][r];
            }
        }
        __syncthreads();   // new h visible
    }

    // ---- epilogue: logits for the final step (t = T-1) from S_T in hB
    if (w < 4) {
        v8f lacc = (v8f){0,0,0,0,0,0,0,0};
        for (int kb = 0; kb < H / 32; ++kb) {
            v16bf a = load_a_frag(hB, kb, lane);
            const v16bf bfv = *(const v16bf*)(wout_pk + ((size_t)w * (H / 32) + kb) * 512 + lane * 16);
            lacc = __builtin_amdgcn_wmma_f32_16x16x32_bf16(
                false, a, false, bfv, (short)0, lacc, false, false);
        }
        int vcol = w * 16 + nn;
        float bo = b_out[vcol];
#pragma unroll
        for (int r = 0; r < 8; ++r) {
            int m  = r + 8 * hi;
            int bg = wg * 16 + m;
            out[((size_t)bg * T + (T - 1)) * V + vcol] = lacc[r] + bo;
        }
    }
}

// ---------------------------------------------------------------------------
// Launch
// ---------------------------------------------------------------------------
extern "C" void kernel_launch(void* const* d_in, const int* in_sizes, int n_in,
                              void* d_out, int out_size, void* d_ws, size_t ws_size,
                              hipStream_t stream) {
    const float* z     = (const float*)d_in[0];
    const float* c     = (const float*)d_in[1];
    const int*   seq   = (const int*)  d_in[2];
    const float* emb   = (const float*)d_in[3];
    const float* W_fch = (const float*)d_in[4];
    const float* b_fch = (const float*)d_in[5];
    const float* W_ih  = (const float*)d_in[6];
    const float* b_ih  = (const float*)d_in[7];
    const float* W_hh  = (const float*)d_in[8];
    const float* b_hh  = (const float*)d_in[9];
    const float* W_out = (const float*)d_in[10];
    const float* b_out = (const float*)d_in[11];
    float* out = (float*)d_out;

    char* ws = (char*)d_ws;
    float* h0f     = (float*)(ws + OFF_H0);
    float* gtab    = (float*)(ws + OFF_GTAB);
    float* gctx    = (float*)(ws + OFF_GCTX);
    bf16*  whh_pk  = (bf16*) (ws + OFF_WHH);
    bf16*  wout_pk = (bf16*) (ws + OFF_WOUT);

    k_h0  <<<(B * H)   / 256, 256, 0, stream>>>(z, c, W_fch, b_fch, h0f);
    k_gtab<<<(V * G3H) / 256, 256, 0, stream>>>(emb, W_ih, gtab);
    k_gctx<<<(B * G3H) / 256, 256, 0, stream>>>(z, c, W_ih, b_ih, gctx);
    // W_hh: 192 nblocks x 32 kblocks x 512 elems
    k_pack<<<((G3H / 16) * (H / 32) * 512) / 256, 256, 0, stream>>>(W_hh, whh_pk, H / 32);
    // W_out: 4 nblocks x 32 kblocks x 512 elems
    k_pack<<<((V / 16) * (H / 32) * 512) / 256, 256, 0, stream>>>(W_out, wout_pk, H / 32);

    gru_scan<<<B / 16, TPB, 0, stream>>>(h0f, gtab, gctx, whh_pk, wout_pk,
                                         seq, b_hh, b_out, out);
}
